// SemanticDecoder_46042049413438
// MI455X (gfx1250) — matrix operand order repaired and checked by reading, not compiled, not run on hardware
//
#include <hip/hip_runtime.h>
#include <hip/hip_bf16.h>
#include <stdint.h>

// ---------------------------------------------------------------------------
// CDNA5 (gfx1250) implementation of the VMamba semantic decoder.
// Dense projections -> bf16 WMMA GEMM (v_wmma_f32_16x16x32_bf16, fp32 acc),
// vectorized b128 global->LDS staging, K padded to multiples of 64 so the
// GEMM has NO scalar fallback path. Selective scan = 4096 channel threads.
// ---------------------------------------------------------------------------

typedef __bf16 bf16_t;
typedef __bf16 v16bf  __attribute__((ext_vector_type(16)));
typedef __bf16 bf16x8 __attribute__((ext_vector_type(8)));
typedef float  v8f    __attribute__((ext_vector_type(8)));

__device__ __forceinline__ bf16_t f2bf(float f) {
  union { float f; uint32_t u; } v; v.f = f;
  uint32_t r = v.u + 0x7FFFu + ((v.u >> 16) & 1u);   // round-to-nearest-even
  uint16_t h = (uint16_t)(r >> 16);
  bf16_t o; __builtin_memcpy(&o, &h, 2); return o;
}
__device__ __forceinline__ float siluf(float x) { return x / (1.f + __expf(-x)); }
__device__ __forceinline__ float geluf(float x) { return 0.5f * x * (1.f + erff(x * 0.70710678118654752f)); }

// ========================== WMMA GEMM ======================================
// C[M,N] = act( (A[M,K] * B[N,K]^T) (+bias) (*scale+shift) ) (+resid)
// REQUIRES: K % 64 == 0, M % 64 == 0 (guaranteed by callers via padding).
// A, B bf16 row-major; lda/ldb multiples of 8 (16B-aligned vector loads).
// 256 threads = 8 waves (2M x 4N); block tile 64x64, K tile 64
// (two K=32 WMMA steps); each wave: 32x16 output = 2 accumulators.
#define BM 64
#define BN 64
#define BKT 64

__global__ __launch_bounds__(256)
void k_gemm_bf16(const bf16_t* __restrict__ A, int lda,
                 const bf16_t* __restrict__ Bw, int ldb,
                 float* __restrict__ C, int ldc,
                 const float* __restrict__ bias,
                 const float* __restrict__ scale,
                 const float* __restrict__ shift,
                 const float* __restrict__ resid,
                 int M, int N, int K, int act)
{
  __shared__ bf16_t As[BM][BKT];
  __shared__ bf16_t Bs[BN][BKT];
  const int tid  = threadIdx.x;
  const int lane = tid & 31;
  const int wave = tid >> 5;            // 0..7
  const int wm = wave >> 2;             // 0..1 : 32-row slab
  const int wn = wave & 3;              // 0..3 : 16-col slab
  const int bM = blockIdx.y * BM;
  const int bN = blockIdx.x * BN;

  v8f acc[2];
  acc[0] = (v8f){0.f, 0.f, 0.f, 0.f, 0.f, 0.f, 0.f, 0.f};
  acc[1] = (v8f){0.f, 0.f, 0.f, 0.f, 0.f, 0.f, 0.f, 0.f};
  bf16x8 zv;
  {
    const bf16_t bz = f2bf(0.f);
    #pragma unroll
    for (int t = 0; t < 8; ++t) zv[t] = bz;
  }

  for (int k0 = 0; k0 < K; k0 += BKT) {
    // Prefetch next K tiles into L2 (global_prefetch_b8)
    if (k0 + BKT < K) {
      __builtin_prefetch(A + (size_t)(bM + (tid >> 2)) * lda + k0 + BKT, 0, 1);
      __builtin_prefetch(Bw + (size_t)(bN + (tid >> 2)) * ldb + k0 + BKT, 0, 1);
    }
    // ---- A & B tiles 64x64: 512 chunks of 8 bf16 each, 2 chunks/thread ----
    #pragma unroll
    for (int j = 0; j < 2; ++j) {
      int chunk = tid + 256 * j;
      int r = chunk >> 3;
      int c = (chunk & 7) * 8;
      // A rows always valid (M % 64 == 0)
      *(bf16x8*)&As[r][c] = *(const bf16x8*)(A + (size_t)(bM + r) * lda + k0 + c);
      // B rows guarded (N may be e.g. 40): vector load or zero vector
      int gr = bN + r;
      bf16x8 bv = zv;
      if (gr < N) bv = *(const bf16x8*)(Bw + (size_t)gr * ldb + k0 + c);
      *(bf16x8*)&Bs[r][c] = bv;
    }
    __syncthreads();

    // ---- two K=32 WMMA steps; B fragment shared by both M subtiles ----
    #pragma unroll
    for (int ks = 0; ks < BKT; ks += 32) {
      v16bf bfrag;
      {
        int col = wn * 16 + (lane & 15);
        int kb2 = ks + (lane >> 4) * 8;
        #pragma unroll
        for (int j = 0; j < 16; ++j) bfrag[j] = Bs[col][kb2 + ((j < 8) ? j : (j + 8))];
      }
      v16bf af0, af1;
      {
        int r0    = wm * 32 + (lane & 15);
        int half8 = (lane >> 4) * 8;
        #pragma unroll
        for (int i = 0; i < 8; ++i) {
          int kb = ks + ((i < 4) ? (2 * i) : (16 + 2 * (i - 4))) + half8;
          af0[2 * i]     = As[r0][kb];
          af0[2 * i + 1] = As[r0][kb + 1];
          af1[2 * i]     = As[r0 + 16][kb];
          af1[2 * i + 1] = As[r0 + 16][kb + 1];
        }
      }
      acc[0] = __builtin_amdgcn_wmma_f32_16x16x32_bf16(false, af0, false, bfrag,
                                                       (short)0, acc[0], false, false);
      acc[1] = __builtin_amdgcn_wmma_f32_16x16x32_bf16(false, af1, false, bfrag,
                                                       (short)0, acc[1], false, false);
    }
    __syncthreads();
  }

  // C/D layout: N = lane&15; M = vgpr_i + 8*(lane>>4). M rows always valid.
  int gn = bN + wn * 16 + (lane & 15);
  if (gn < N) {
    float bsv = bias ? bias[gn] : 0.f;
    float scv = scale ? scale[gn] : 1.f;
    float shv = scale ? shift[gn] : 0.f;
    #pragma unroll
    for (int t = 0; t < 2; ++t) {
      #pragma unroll
      for (int i = 0; i < 8; ++i) {
        int gm = bM + wm * 32 + t * 16 + i + 8 * (lane >> 4);
        float x = acc[t][i];
        if (bias)  x += bsv;
        if (scale) x = x * scv + shv;
        if (act == 1) x = fmaxf(x, 0.f);
        else if (act == 2) x = siluf(x);
        else if (act == 3) x = geluf(x);
        if (resid) x += resid[(size_t)gm * ldc + gn];
        C[(size_t)gm * ldc + gn] = x;
      }
    }
  }
}

// ========================== support kernels ================================

__global__ void k_f32_to_bf16(const float* __restrict__ x, bf16_t* __restrict__ y, size_t n) {
  size_t i = (size_t)blockIdx.x * blockDim.x + threadIdx.x;
  if (i < n) y[i] = f2bf(x[i]);
}

// rows x Kp bf16 output; copy first Kcopy cols of each fp32 row (stride ldx),
// zero-pad cols [Kcopy, Kp).
__global__ void k_cvt_pad(const float* __restrict__ x, int ldx, int Kcopy,
                          bf16_t* __restrict__ y, int Kp, int rows) {
  size_t idx = (size_t)blockIdx.x * blockDim.x + threadIdx.x;
  size_t total = (size_t)rows * Kp;
  if (idx >= total) return;
  int c = (int)(idx % Kp);
  size_t r = idx / Kp;
  y[idx] = (c < Kcopy) ? f2bf(x[r * ldx + c]) : f2bf(0.f);
}

__global__ void k_nchw_to_nlc(const float* __restrict__ x, float* __restrict__ y,
                              int B, int C, int HW, int ldy, int c0) {
  size_t idx = (size_t)blockIdx.x * blockDim.x + threadIdx.x;
  size_t total = (size_t)B * C * HW;
  if (idx >= total) return;
  int p = (int)(idx % HW);
  int c = (int)((idx / HW) % C);
  int b = (int)(idx / ((size_t)HW * C));
  y[((size_t)b * HW + p) * ldy + c0 + c] = x[idx];
}

__global__ void k_nlc_to_nchw(const float* __restrict__ x, float* __restrict__ y,
                              int B, int C, int HW, int ldx, int c0) {
  size_t idx = (size_t)blockIdx.x * blockDim.x + threadIdx.x;
  size_t total = (size_t)B * C * HW;
  if (idx >= total) return;
  int p = (int)(idx % HW);
  int c = (int)((idx / HW) % C);
  int b = (int)(idx / ((size_t)HW * C));
  y[idx] = x[((size_t)b * HW + p) * ldx + c0 + c];
}

__global__ void k_layernorm(const float* __restrict__ x, const float* __restrict__ g,
                            const float* __restrict__ b, float* __restrict__ y, int C) {
  __shared__ float ssum[256], ssq[256];
  int row = blockIdx.x;
  const float* xr = x + (size_t)row * C;
  float s = 0.f, q = 0.f;
  for (int c = threadIdx.x; c < C; c += blockDim.x) { float v = xr[c]; s += v; q += v * v; }
  ssum[threadIdx.x] = s; ssq[threadIdx.x] = q;
  __syncthreads();
  for (int o = blockDim.x >> 1; o > 0; o >>= 1) {
    if ((int)threadIdx.x < o) { ssum[threadIdx.x] += ssum[threadIdx.x + o]; ssq[threadIdx.x] += ssq[threadIdx.x + o]; }
    __syncthreads();
  }
  float mu  = ssum[0] / C;
  float var = ssq[0] / C - mu * mu;
  float inv = rsqrtf(var + 1e-5f);
  float* yr = y + (size_t)row * C;
  for (int c = threadIdx.x; c < C; c += blockDim.x)
    yr[c] = (xr[c] - mu) * inv * g[c] + b[c];
}

__global__ void k_bn_prep(const float* __restrict__ g, const float* __restrict__ b,
                          const float* __restrict__ m, const float* __restrict__ v,
                          const float* __restrict__ convb,
                          float* __restrict__ scale, float* __restrict__ shift, int C) {
  int c = blockIdx.x * blockDim.x + threadIdx.x;
  if (c >= C) return;
  float s = g[c] * rsqrtf(v[c] + 1e-5f);
  scale[c] = s;
  float cb = convb ? convb[c] : 0.f;
  shift[c] = b[c] + (cb - m[c]) * s;
}

__global__ void k_dwconv3x3_silu(const float* __restrict__ x, const float* __restrict__ w9,
                                 const float* __restrict__ bias, float* __restrict__ y,
                                 int B, int C, int H, int W) {
  size_t idx = (size_t)blockIdx.x * blockDim.x + threadIdx.x;
  size_t total = (size_t)B * C * H * W;
  if (idx >= total) return;
  int wx = (int)(idx % W);
  int hy = (int)((idx / W) % H);
  int c  = (int)((idx / ((size_t)W * H)) % C);
  int b  = (int)(idx / ((size_t)W * H * C));
  const float* xp = x + ((size_t)b * C + c) * H * W;
  const float* wc = w9 + (size_t)c * 9;
  float acc = bias[c];
  #pragma unroll
  for (int dy = 0; dy < 3; ++dy) {
    int hh = hy + dy - 1;
    if (hh < 0 || hh >= H) continue;
    #pragma unroll
    for (int dx = 0; dx < 3; ++dx) {
      int ww = wx + dx - 1;
      if (ww < 0 || ww >= W) continue;
      acc += xp[(size_t)hh * W + ww] * wc[dy * 3 + dx];
    }
  }
  y[idx] = siluf(acc);
}

__global__ void k_conv3x3_affine(const float* __restrict__ x, const float* __restrict__ w,
                                 float* __restrict__ y,
                                 const float* __restrict__ scale, const float* __restrict__ shift,
                                 const float* __restrict__ resid, int relu,
                                 int B, int Cin, int Cout, int H, int W) {
  size_t idx = (size_t)blockIdx.x * blockDim.x + threadIdx.x;
  size_t total = (size_t)B * Cout * H * W;
  if (idx >= total) return;
  int wx = (int)(idx % W);
  int hy = (int)((idx / W) % H);
  int co = (int)((idx / ((size_t)W * H)) % Cout);
  int b  = (int)(idx / ((size_t)W * H * Cout));
  const float* wr = w + (size_t)co * Cin * 9;
  float acc = 0.f;
  for (int ci = 0; ci < Cin; ++ci) {
    const float* xp = x + ((size_t)b * Cin + ci) * H * W;
    const float* wc = wr + ci * 9;
    #pragma unroll
    for (int dy = 0; dy < 3; ++dy) {
      int hh = hy + dy - 1;
      if (hh < 0 || hh >= H) continue;
      #pragma unroll
      for (int dx = 0; dx < 3; ++dx) {
        int ww = wx + dx - 1;
        if (ww < 0 || ww >= W) continue;
        acc += xp[(size_t)hh * W + ww] * wc[dy * 3 + dx];
      }
    }
  }
  float v = acc;
  if (scale) v = v * scale[co] + shift[co];
  if (resid) v += resid[idx];
  if (relu)  v = fmaxf(v, 0.f);
  y[idx] = v;
}

// Build the 4 directional scan sequences as (4, B*L, D) fp32 + bf16.
__global__ void k_build_xs(const float* __restrict__ xc,      // (B,256,H,W)
                           float* __restrict__ xs_f, bf16_t* __restrict__ xs_bf,
                           int B, int H, int W) {
  const int D = 256;
  int L = H * W;
  size_t idx = (size_t)blockIdx.x * blockDim.x + threadIdx.x;
  size_t total = (size_t)B * D * L;
  if (idx >= total) return;
  int p = (int)(idx % L);
  int d = (int)((idx / L) % D);
  int b = (int)(idx / ((size_t)L * D));
  int h = p / W, w = p % W;
  float v = xc[idx];
  bf16_t vb = f2bf(v);
  int l1 = w * H + h;
  size_t TL = (size_t)B * L;
  size_t r0 = ((size_t)0 * TL + (size_t)b * L + p)            * D + d;
  size_t r1 = ((size_t)1 * TL + (size_t)b * L + l1)           * D + d;
  size_t r2 = ((size_t)2 * TL + (size_t)b * L + (L - 1 - p))  * D + d;
  size_t r3 = ((size_t)3 * TL + (size_t)b * L + (L - 1 - l1)) * D + d;
  xs_f[r0] = v; xs_f[r1] = v; xs_f[r2] = v; xs_f[r3] = v;
  xs_bf[r0] = vb; xs_bf[r1] = vb; xs_bf[r2] = vb; xs_bf[r3] = vb;
}

// Sequential SSM per (k,b,d) channel; dtv buffer is read (delta) then
// overwritten in place with y (same element, read-before-write per thread).
__global__ void k_selective_scan(const float* __restrict__ xs,      // (4,BL,256)
                                 float* dtv_ys,                     // in: dt, out: y
                                 const float* __restrict__ xdbl,    // (4,BL,40): B@8, C@24
                                 const float* __restrict__ A_log,   // (4,256,16)
                                 const float* __restrict__ Ds,      // (4,256)
                                 const float* __restrict__ dt_bias, // (4,256)
                                 int B, int L) {
  const int D = 256, N = 16;
  int idx = blockIdx.x * blockDim.x + threadIdx.x;
  if (idx >= 4 * B * D) return;
  int d = idx % D;
  int b = (idx / D) % B;
  int k = idx / (D * B);
  float Ac[N];
  #pragma unroll
  for (int n = 0; n < N; ++n) Ac[n] = -__expf(A_log[((size_t)k * D + d) * N + n]);
  float bias = dt_bias[k * D + d];
  float Dd   = Ds[k * D + d];
  float h[N];
  #pragma unroll
  for (int n = 0; n < N; ++n) h[n] = 0.f;
  size_t base = ((size_t)k * B + b) * (size_t)L;
  for (int l = 0; l < L; ++l) {
    size_t row = base + l;
    float dt = dtv_ys[row * D + d] + bias;
    dt = (dt > 20.f) ? dt : log1pf(__expf(dt));          // softplus
    float u  = xs[row * D + d];
    float du = dt * u;
    const float* Bv = xdbl + row * 40 + 8;
    const float* Cv = xdbl + row * 40 + 24;
    float y = 0.f;
    #pragma unroll
    for (int n = 0; n < N; ++n) {
      h[n] = h[n] * __expf(dt * Ac[n]) + du * Bv[n];
      y += h[n] * Cv[n];
    }
    dtv_ys[row * D + d] = y + Dd * u;
  }
}

// Merge 4 directional scan outputs back into spatial token order (B*L, 256).
__global__ void k_merge_y(const float* __restrict__ ys, float* __restrict__ out,
                          int B, int H, int W) {
  const int D = 256;
  int L = H * W;
  size_t idx = (size_t)blockIdx.x * blockDim.x + threadIdx.x;
  size_t total = (size_t)B * L * D;
  if (idx >= total) return;
  int d = (int)(idx % D);
  int p = (int)((idx / D) % L);
  int b = (int)(idx / ((size_t)D * L));
  int h = p / W, w = p % W;
  int l1 = w * H + h;
  size_t TL = (size_t)B * L;
  float s = ys[((size_t)0 * TL + (size_t)b * L + p)            * D + d]
          + ys[((size_t)1 * TL + (size_t)b * L + l1)           * D + d]
          + ys[((size_t)2 * TL + (size_t)b * L + (L - 1 - p))  * D + d]
          + ys[((size_t)3 * TL + (size_t)b * L + (L - 1 - l1)) * D + d];
  out[idx] = s;
}

__global__ void k_mul_silu_bf16(const float* __restrict__ x, const float* __restrict__ z,
                                int zld, int zoff, bf16_t* __restrict__ out,
                                int rows, int C) {
  size_t idx = (size_t)blockIdx.x * blockDim.x + threadIdx.x;
  size_t total = (size_t)rows * C;
  if (idx >= total) return;
  int c = (int)(idx % C);
  size_t r = idx / C;
  out[idx] = f2bf(x[idx] * siluf(z[r * zld + zoff + c]));
}

// out(B,C,2Hs,2Ws) = bilinear_upsample_2x(x) + y  (half-pixel centers)
__global__ void k_up2x_add(const float* __restrict__ x, const float* __restrict__ y,
                           float* __restrict__ out, int B, int C, int Hs, int Ws) {
  int Hd = Hs * 2, Wd = Ws * 2;
  size_t idx = (size_t)blockIdx.x * blockDim.x + threadIdx.x;
  size_t total = (size_t)B * C * Hd * Wd;
  if (idx >= total) return;
  int wd = (int)(idx % Wd);
  int hd = (int)((idx / Wd) % Hd);
  int c  = (int)((idx / ((size_t)Wd * Hd)) % C);
  int b  = (int)(idx / ((size_t)Wd * Hd * C));
  float sy = (hd + 0.5f) * 0.5f - 0.5f;
  float sx = (wd + 0.5f) * 0.5f - 0.5f;
  int y0 = (int)floorf(sy), x0 = (int)floorf(sx);
  float fy = sy - y0, fx = sx - x0;
  int y0c = min(max(y0, 0), Hs - 1), y1c = min(max(y0 + 1, 0), Hs - 1);
  int x0c = min(max(x0, 0), Ws - 1), x1c = min(max(x0 + 1, 0), Ws - 1);
  const float* xp = x + ((size_t)b * C + c) * Hs * Ws;
  float v00 = xp[(size_t)y0c * Ws + x0c], v01 = xp[(size_t)y0c * Ws + x1c];
  float v10 = xp[(size_t)y1c * Ws + x0c], v11 = xp[(size_t)y1c * Ws + x1c];
  float v = v00 * (1 - fy) * (1 - fx) + v01 * (1 - fy) * fx
          + v10 * fy * (1 - fx)       + v11 * fy * fx;
  out[idx] = v + y[idx];
}

// ========================== host orchestration =============================

static inline dim3 g1d(size_t n) { return dim3((unsigned)((n + 255) / 256)); }

struct VssP {
  const float *A_log, *Ds, *conv_b, *conv_w, *dt_b, *dt_w,
              *fc1_b, *fc1_w, *fc2_b, *fc2_w, *in_proj_w,
              *ln1_b, *ln1_g, *ln2_b, *ln2_g,
              *out_norm_b, *out_norm_g, *out_proj_w, *x_proj_w;
};
struct ResP {
  const float *bn1_b, *bn1_g, *bn1_m, *bn1_v;
  const float *bn2_b, *bn2_g, *bn2_m, *bn2_v;
  const float *c1_w, *c2_w;
};
struct TransP {
  const float *b;
  const float *bn_b, *bn_g, *bn_m, *bn_v;
  const float *w;
};

// Flattening order assumption: jax tree flatten = dict keys sorted.
static VssP readVss(void* const* d, int& i) {
  VssP p;
  p.A_log = (const float*)d[i++]; p.Ds = (const float*)d[i++];
  p.conv_b = (const float*)d[i++]; p.conv_w = (const float*)d[i++];
  p.dt_b = (const float*)d[i++]; p.dt_w = (const float*)d[i++];
  p.fc1_b = (const float*)d[i++]; p.fc1_w = (const float*)d[i++];
  p.fc2_b = (const float*)d[i++]; p.fc2_w = (const float*)d[i++];
  p.in_proj_w = (const float*)d[i++];
  p.ln1_b = (const float*)d[i++]; p.ln1_g = (const float*)d[i++];
  p.ln2_b = (const float*)d[i++]; p.ln2_g = (const float*)d[i++];
  p.out_norm_b = (const float*)d[i++]; p.out_norm_g = (const float*)d[i++];
  p.out_proj_w = (const float*)d[i++]; p.x_proj_w = (const float*)d[i++];
  return p;
}
static ResP readRes(void* const* d, int& i) {
  ResP p;
  p.bn1_b = (const float*)d[i++]; p.bn1_g = (const float*)d[i++];
  p.bn1_m = (const float*)d[i++]; p.bn1_v = (const float*)d[i++];
  p.bn2_b = (const float*)d[i++]; p.bn2_g = (const float*)d[i++];
  p.bn2_m = (const float*)d[i++]; p.bn2_v = (const float*)d[i++];
  p.c1_w = (const float*)d[i++]; p.c2_w = (const float*)d[i++];
  return p;
}
static TransP readTrans(void* const* d, int& i) {
  TransP p;
  p.b = (const float*)d[i++];
  p.bn_b = (const float*)d[i++]; p.bn_g = (const float*)d[i++];
  p.bn_m = (const float*)d[i++]; p.bn_v = (const float*)d[i++];
  p.w = (const float*)d[i++];
  return p;
}

struct Ws {
  float *t_h, *t_h2, *t_tok, *t_tok2, *t_tok3;
  float *xs_f, *xdbl, *dtv;
  float *nchwA, *nchwB;
  float *g0, *g1, *g2, *g3, *g4;
  float *sc1, *sh1, *sc2, *sh2;
  bf16_t *bfA, *bfB, *bfW, *bfXs, *bfXd;
};

struct Bump {
  char* base; size_t off;
  template <class T> T* take(size_t n) {
    size_t bytes = (n * sizeof(T) + 255) & ~(size_t)255;
    T* p = (T*)(base + off); off += bytes; return p;
  }
};

static void setupWs(Ws& w, void* d_ws) {
  Bump a{(char*)d_ws, 0};
  const size_t T = 16384;   // max tokens (B=4, 64x64)
  w.t_h    = a.take<float>(T * 128);
  w.t_h2   = a.take<float>(T * 128);
  w.t_tok  = a.take<float>(T * 512);
  w.t_tok2 = a.take<float>(T * 256);
  w.t_tok3 = a.take<float>(T * 256);
  w.xs_f   = a.take<float>(4 * T * 256);
  w.xdbl   = a.take<float>(4 * T * 40);
  w.dtv    = a.take<float>(4 * T * 256);   // reused in-place as ys
  w.nchwA  = a.take<float>((size_t)4 * 256 * 64 * 64);
  w.nchwB  = a.take<float>((size_t)4 * 256 * 64 * 64);
  w.g0 = a.take<float>((size_t)4 * 128 * 64 * 64);
  w.g1 = a.take<float>((size_t)4 * 128 * 64 * 64);
  w.g2 = a.take<float>((size_t)4 * 128 * 64 * 64);
  w.g3 = a.take<float>((size_t)4 * 128 * 64 * 64);
  w.g4 = a.take<float>((size_t)4 * 128 * 64 * 64);
  w.sc1 = a.take<float>(1024); w.sh1 = a.take<float>(1024);
  w.sc2 = a.take<float>(1024); w.sh2 = a.take<float>(1024);
  w.bfA  = a.take<bf16_t>(T * 256);
  w.bfB  = a.take<bf16_t>(T * 512);
  w.bfW  = a.take<bf16_t>(262144);
  w.bfXs = a.take<bf16_t>(4 * T * 256);
  w.bfXd = a.take<bf16_t>(4 * T * 64);     // dt A operand, padded K=64
}

static void cvt(hipStream_t s, const float* x, bf16_t* y, size_t n) {
  k_f32_to_bf16<<<g1d(n), 256, 0, s>>>(x, y, n);
}
static void cvt_pad(hipStream_t s, const float* x, int ldx, int Kcopy,
                    bf16_t* y, int Kp, int rows) {
  k_cvt_pad<<<g1d((size_t)rows * Kp), 256, 0, s>>>(x, ldx, Kcopy, y, Kp, rows);
}

static void gemm(hipStream_t s, const bf16_t* A, int lda, const bf16_t* Bw, int ldb,
                 float* C, int ldc, const float* bias, const float* scale,
                 const float* shift, const float* resid,
                 int M, int N, int K, int act) {
  dim3 grid((N + BN - 1) / BN, (M + BM - 1) / BM);
  k_gemm_bf16<<<grid, 256, 0, s>>>(A, lda, Bw, ldb, C, ldc, bias, scale, shift,
                                   resid, M, N, K, act);
}

// 1x1 conv over channels via WMMA GEMM (NCHW -> NLC -> GEMM -> NCHW).
// K padded up to a multiple of 64 (Cin=96 -> 128).
static void run_1x1(hipStream_t s, Ws& w, const float* feat, int Cin,
                    const float* wgt, const float* bias,
                    const float* scale, const float* shift, int relu,
                    float* out, int B, int H, int W) {
  int L = H * W, T = B * L;
  int Kp = (Cin + 63) & ~63;
  k_nchw_to_nlc<<<g1d((size_t)T * Cin), 256, 0, s>>>(feat, w.t_tok, B, Cin, L, Cin, 0);
  cvt_pad(s, w.t_tok, Cin, Cin, w.bfB, Kp, T);
  cvt_pad(s, wgt, Cin, Cin, w.bfW, Kp, 128);
  gemm(s, w.bfB, Kp, w.bfW, Kp, w.t_tok2, 128, bias, scale, shift, nullptr,
       T, 128, Kp, relu);
  k_nlc_to_nchw<<<g1d((size_t)T * 128), 256, 0, s>>>(w.t_tok2, out, B, 128, L, 128, 0);
}

static void run_trans(hipStream_t s, Ws& w, const TransP& P, const float* feat,
                      int Cin, float* out, int B, int H, int W) {
  k_bn_prep<<<1, 128, 0, s>>>(P.bn_g, P.bn_b, P.bn_m, P.bn_v, P.b, w.sc1, w.sh1, 128);
  run_1x1(s, w, feat, Cin, P.w, nullptr, w.sc1, w.sh1, /*relu=*/1, out, B, H, W);
}

static void run_res(hipStream_t s, Ws& w, const ResP& P, const float* x,
                    float* out, float* tmp, int B, int H, int W) {
  k_bn_prep<<<1, 128, 0, s>>>(P.bn1_g, P.bn1_b, P.bn1_m, P.bn1_v, nullptr, w.sc1, w.sh1, 128);
  k_bn_prep<<<1, 128, 0, s>>>(P.bn2_g, P.bn2_b, P.bn2_m, P.bn2_v, nullptr, w.sc2, w.sh2, 128);
  size_t n = (size_t)B * 128 * H * W;
  k_conv3x3_affine<<<g1d(n), 256, 0, s>>>(x, P.c1_w, tmp, w.sc1, w.sh1, nullptr, 1,
                                          B, 128, 128, H, W);
  k_conv3x3_affine<<<g1d(n), 256, 0, s>>>(tmp, P.c2_w, out, w.sc2, w.sh2, x, 1,
                                          B, 128, 128, H, W);
}

static void run_vss(hipStream_t s, Ws& w, const VssP& P, const float* x_nchw,
                    float* out_nchw, int B, int H, int W) {
  const int C = 128, D = 256, L = H * W, T = B * L;

  // h tokens + LN1
  k_nchw_to_nlc<<<g1d((size_t)T * C), 256, 0, s>>>(x_nchw, w.t_h, B, C, L, C, 0);
  k_layernorm<<<T, 256, 0, s>>>(w.t_h, P.ln1_g, P.ln1_b, w.t_tok2, C);

  // in_proj: (T,128) -> (T,512) [xx | z]   (K=128)
  cvt(s, w.t_tok2, w.bfA, (size_t)T * C);
  cvt(s, P.in_proj_w, w.bfW, (size_t)512 * 128);
  gemm(s, w.bfA, 128, w.bfW, 128, w.t_tok, 512, nullptr, nullptr, nullptr, nullptr,
       T, 512, 128, 0);

  // depthwise conv + SiLU on xx
  k_nlc_to_nchw<<<g1d((size_t)T * D), 256, 0, s>>>(w.t_tok, w.nchwA, B, D, L, 512, 0);
  k_dwconv3x3_silu<<<g1d((size_t)B * D * L), 256, 0, s>>>(w.nchwA, P.conv_w, P.conv_b,
                                                          w.nchwB, B, D, H, W);
  // 4-direction sequences
  k_build_xs<<<g1d((size_t)B * D * L), 256, 0, s>>>(w.nchwB, w.xs_f, w.bfXs, B, H, W);

  // x_proj: per-direction (T,256) -> (T,40)   (K=256)
  cvt(s, P.x_proj_w, w.bfW, (size_t)4 * 40 * 256);
  for (int k = 0; k < 4; ++k)
    gemm(s, w.bfXs + (size_t)k * T * 256, 256, w.bfW + (size_t)k * 40 * 256, 256,
         w.xdbl + (size_t)k * T * 40, 40, nullptr, nullptr, nullptr, nullptr,
         T, 40, 256, 0);

  // dt projection: per-direction (T,8) -> (T,256), K padded 8 -> 64
  cvt_pad(s, w.xdbl, 40, 8, w.bfXd, 64, 4 * T);       // dts cols only, zero-pad
  cvt_pad(s, P.dt_w, 8, 8, w.bfW, 64, 4 * 256);       // (4*256, 8) -> (4*256, 64)
  for (int k = 0; k < 4; ++k)
    gemm(s, w.bfXd + (size_t)k * T * 64, 64, w.bfW + (size_t)k * 256 * 64, 64,
         w.dtv + (size_t)k * T * 256, 256, nullptr, nullptr, nullptr, nullptr,
         T, 256, 64, 0);

  // selective scan (y written in place over dtv)
  k_selective_scan<<<g1d((size_t)4 * B * D), 256, 0, s>>>(w.xs_f, w.dtv, w.xdbl,
                                                          P.A_log, P.Ds, P.dt_b, B, L);

  // merge directions, out-norm, gate by SiLU(z)
  k_merge_y<<<g1d((size_t)T * D), 256, 0, s>>>(w.dtv, w.t_tok2, B, H, W);
  k_layernorm<<<T, 256, 0, s>>>(w.t_tok2, P.out_norm_g, P.out_norm_b, w.t_tok3, D);
  k_mul_silu_bf16<<<g1d((size_t)T * D), 256, 0, s>>>(w.t_tok3, w.t_tok, 512, 256,
                                                     w.bfA, T, D);

  // out_proj + residual h   (K=256)
  cvt(s, P.out_proj_w, w.bfW, (size_t)128 * 256);
  gemm(s, w.bfA, 256, w.bfW, 256, w.t_h2, 128, nullptr, nullptr, nullptr, w.t_h,
       T, 128, 256, 0);

  // MLP: LN2 -> fc1(GELU, K=128) -> fc2(K=512) + residual
  k_layernorm<<<T, 256, 0, s>>>(w.t_h2, P.ln2_g, P.ln2_b, w.t_tok2, C);
  cvt(s, w.t_tok2, w.bfA, (size_t)T * C);
  cvt(s, P.fc1_w, w.bfW, (size_t)512 * 128);
  gemm(s, w.bfA, 128, w.bfW, 128, w.t_tok, 512, P.fc1_b, nullptr, nullptr, nullptr,
       T, 512, 128, 3 /*gelu*/);
  cvt(s, w.t_tok, w.bfB, (size_t)T * 512);
  cvt(s, P.fc2_w, w.bfW, (size_t)128 * 512);
  gemm(s, w.bfB, 512, w.bfW, 512, w.t_h, 128, P.fc2_b, nullptr, nullptr, w.t_h2,
       T, 128, 512, 0);

  k_nlc_to_nchw<<<g1d((size_t)T * C), 256, 0, s>>>(w.t_h, out_nchw, B, C, L, 128, 0);
}

extern "C" void kernel_launch(void* const* d_in, const int* in_sizes, int n_in,
                              void* d_out, int out_size, void* d_ws, size_t ws_size,
                              hipStream_t stream) {
  (void)in_sizes; (void)n_in; (void)out_size; (void)ws_size;
  int ii = 0;
  const float* feat1 = (const float*)d_in[ii++];
  const float* feat2 = (const float*)d_in[ii++];
  const float* feat3 = (const float*)d_in[ii++];
  const float* feat4 = (const float*)d_in[ii++];
  VssP b1 = readVss(d_in, ii);
  VssP b2 = readVss(d_in, ii);
  VssP b3 = readVss(d_in, ii);
  VssP b4 = readVss(d_in, ii);
  const float* c4_b = (const float*)d_in[ii++];
  const float* c4_w = (const float*)d_in[ii++];
  ResP s0 = readRes(d_in, ii);
  ResP s1 = readRes(d_in, ii);
  ResP s2 = readRes(d_in, ii);
  ResP s3 = readRes(d_in, ii);
  TransP t1 = readTrans(d_in, ii);
  TransP t2 = readTrans(d_in, ii);
  TransP t3 = readTrans(d_in, ii);

  Ws w; setupWs(w, d_ws);
  const int B = 4;

  // p4 = VSS(conv1x1(feat4))
  run_1x1(stream, w, feat4, 768, c4_w, c4_b, nullptr, nullptr, 0, w.g1, B, 8, 8);
  run_vss(stream, w, b4, w.g1, w.g0, B, 8, 8);

  // level 3: 16x16
  run_trans(stream, w, t3, feat3, 384, w.g1, B, 16, 16);
  k_up2x_add<<<g1d((size_t)B * 128 * 16 * 16), 256, 0, stream>>>(w.g0, w.g1, w.g2, B, 128, 8, 8);
  run_res(stream, w, s3, w.g2, w.g3, w.g4, B, 16, 16);
  run_vss(stream, w, b3, w.g3, w.g0, B, 16, 16);

  // level 2: 32x32
  run_trans(stream, w, t2, feat2, 192, w.g1, B, 32, 32);
  k_up2x_add<<<g1d((size_t)B * 128 * 32 * 32), 256, 0, stream>>>(w.g0, w.g1, w.g2, B, 128, 16, 16);
  run_res(stream, w, s2, w.g2, w.g3, w.g4, B, 32, 32);
  run_vss(stream, w, b2, w.g3, w.g0, B, 32, 32);

  // level 1: 64x64
  run_trans(stream, w, t1, feat1, 96, w.g1, B, 64, 64);
  k_up2x_add<<<g1d((size_t)B * 128 * 64 * 64), 256, 0, stream>>>(w.g0, w.g1, w.g2, B, 128, 32, 32);
  run_res(stream, w, s1, w.g2, w.g3, w.g4, B, 64, 64);
  run_vss(stream, w, b1, w.g3, w.g0, B, 64, 64);

  // final residual block -> d_out (B,128,64,64) fp32
  run_res(stream, w, s0, w.g0, (float*)d_out, w.g4, B, 64, 64);
}
